// MultiHeadedAttention_45260365365792
// MI455X (gfx1250) — matrix operand order repaired and checked by reading, not compiled, not run on hardware
//
#include <hip/hip_runtime.h>
#include <hip/hip_bf16.h>

// ---------------------------------------------------------------------------
// MI455X (gfx1250, wave32) implementation.
// GEMMs use v_wmma_f32_16x16x32_f16 (f16 inputs, f32 accumulate).
// Scores-GEMM tiles are staged with global_load_async_to_lds_b128 (ASYNCcnt)
// since that stage is a pure f16 copy. Conv intermediate stored f16 to halve
// HBM traffic; conv2 fused with the final attn@value reduction so the second
// [4,8,900,900] tensor is never materialized.
// ---------------------------------------------------------------------------

typedef __attribute__((ext_vector_type(16))) _Float16 v16h;
typedef __attribute__((ext_vector_type(8)))  _Float16 v8h;
typedef __attribute__((ext_vector_type(8)))  float    v8f;

// Async 16-byte global -> LDS copy (gfx1250, tracked by ASYNCcnt).
// vdst VGPR holds the wave-relative LDS byte address (addrspace(3) ptr).
__device__ inline void async_ld16(void* lds, const void* g) {
  asm volatile("global_load_async_to_lds_b128 %0, %1, off"
               :: "v"((__attribute__((address_space(3))) void*)lds), "v"(g)
               : "memory");
}
__device__ inline void wait_async0() {
  asm volatile("s_wait_asynccnt 0x0" ::: "memory");
}

// A-fragment (16x32 f16, MxK) from LDS tile stored row-major [rows][32].
// Lane L: m = L&15, g = L>>4; element e -> k = g*16 + e (contiguous).
__device__ inline v16h load_frag_a(const _Float16* As, int row_base, int lane) {
  int g = lane >> 4, m = lane & 15;
  const _Float16* r = As + (size_t)(row_base + m) * 32;
  v8h lo = *(const v8h*)(r + g * 16);
  v8h hi = *(const v8h*)(r + g * 16 + 8);
  return __builtin_shufflevector(lo, hi, 0, 1, 2, 3, 4, 5, 6, 7,
                                 8, 9, 10, 11, 12, 13, 14, 15);
}

// B-fragment (32x16 f16, KxN) from LDS tile stored TRANSPOSED: Bt[n][k], 32 k
// per row. Lane L: n = L&15, g = L>>4; element e -> k = g*16 + e (contiguous).
__device__ inline v16h load_frag_b(const _Float16* Bt, int row_base, int lane) {
  int g = lane >> 4, n = lane & 15;
  const _Float16* r = Bt + (size_t)(row_base + n) * 32;
  return *(const v16h*)(r + g * 16);
}

// ---------------------------------------------------------------------------
// Kernel 1: Y = X @ W^T + bias, output f16 in per-head layout
//   X: [3600, 512] f32 (query or key_t flattened), W: [512, 512] f32 (row = out)
//   Y[( (b*8+h)*900 + p )*64 + c], n = h*64+c, m = b*900+p
// Block: 256 thr (8 waves), tile M=16 x N=128, K-loop 512/32.
// (f32 -> f16 conversion happens during staging, so this stage keeps VGPR
//  copies rather than async.)
// ---------------------------------------------------------------------------
__global__ __launch_bounds__(256) void proj_kernel(
    const float* __restrict__ X, const float* __restrict__ W,
    const float* __restrict__ bias, _Float16* __restrict__ Y) {
  __shared__ __attribute__((aligned(32))) _Float16 As[16 * 32];
  __shared__ __attribute__((aligned(32))) _Float16 Bs[128 * 32];
  int tid = threadIdx.x, lane = tid & 31, wave = tid >> 5;
  int bm = blockIdx.x * 16;    // 225 blocks
  int bn = blockIdx.y * 128;   // 4 blocks
  v8f acc = {};
  for (int k0 = 0; k0 < 512; k0 += 32) {
    __syncthreads();
    for (int i = tid; i < 16 * 32; i += 256) {
      int r = i >> 5, c = i & 31;
      As[i] = (_Float16)X[(size_t)(bm + r) * 512 + k0 + c];
    }
    for (int i = tid; i < 128 * 32; i += 256) {
      int r = i >> 5, c = i & 31;
      Bs[i] = (_Float16)W[(size_t)(bn + r) * 512 + k0 + c];  // Bt[n][k]=W[n][k]
    }
    __syncthreads();
    v16h a = load_frag_a(As, 0, lane);
    v16h b = load_frag_b(Bs, wave * 16, lane);
    acc = __builtin_amdgcn_wmma_f32_16x16x32_f16(false, a, false, b,
                                                 (short)0, acc, false, false);
  }
  int g = lane >> 4, nn = lane & 15;
  int n = bn + wave * 16 + nn;
  int h = n >> 6, cc = n & 63;
  float bv = bias[n];
#pragma unroll
  for (int r = 0; r < 8; ++r) {
    int m = bm + r + 8 * g;
    int b_ = m / 900, p = m % 900;
    Y[((size_t)(b_ * 8 + h) * 900 + p) * 64 + cc] = (_Float16)(acc[r] + bv);
  }
}

// ---------------------------------------------------------------------------
// Kernel 2: scores[bh][p][q] = sum_c qh[bh][p][c] * kh[bh][q][c]  (f32 out)
// Block: 256 thr (8 waves, 2x4), tile M=32 x N=64, K=64 (2 iters).
// Staging uses async global->LDS B128 copies; edge rows are CLAMPED to 899
// instead of zero-filled (duplicated rows only feed masked-out outputs).
// ---------------------------------------------------------------------------
__global__ __launch_bounds__(256) void scores_kernel(
    const _Float16* __restrict__ qh, const _Float16* __restrict__ kh,
    float* __restrict__ scores) {
  __shared__ __attribute__((aligned(32))) _Float16 As[32 * 32];
  __shared__ __attribute__((aligned(32))) _Float16 Bs[64 * 32];
  int tid = threadIdx.x, lane = tid & 31, wave = tid >> 5;
  int wm = wave >> 2, wn = wave & 3;
  int bm = blockIdx.x * 32, bn = blockIdx.y * 64, bh = blockIdx.z;
  const _Float16* qb = qh + (size_t)bh * 900 * 64;
  const _Float16* kb = kh + (size_t)bh * 900 * 64;
  float* sb = scores + (size_t)bh * 810000;
  v8f acc = {};
  for (int k0 = 0; k0 < 64; k0 += 32) {
    __syncthreads();
    // 384 x 16B chunks: As = 128 chunks (32 rows x 64B), Bs = 256 chunks.
    for (int c = tid; c < 384; c += 256) {
      if (c < 128) {
        int r = c >> 2, sub = c & 3;
        int m = bm + r;
        if (m > 899) m = 899;  // clamp: duplicates only feed masked outputs
        async_ld16(&As[c * 8], qb + (size_t)m * 64 + k0 + sub * 8);
      } else {
        int cb = c - 128;
        int r = cb >> 2, sub = cb & 3;
        int n = bn + r;
        if (n > 899) n = 899;
        async_ld16(&Bs[cb * 8], kb + (size_t)n * 64 + k0 + sub * 8);
      }
    }
    wait_async0();
    __syncthreads();
    v16h a = load_frag_a(As, wm * 16, lane);
    v16h b = load_frag_b(Bs, wn * 16, lane);
    acc = __builtin_amdgcn_wmma_f32_16x16x32_f16(false, a, false, b,
                                                 (short)0, acc, false, false);
  }
  int g = lane >> 4, nn = lane & 15;
  int gn = bn + wn * 16 + nn;
#pragma unroll
  for (int r = 0; r < 8; ++r) {
    int gm = bm + wm * 16 + r + 8 * g;
    if (gm < 900 && gn < 900) sb[(size_t)gm * 900 + gn] = acc[r];
  }
}

// ---------------------------------------------------------------------------
// Kernel 3: idx[bh][q] = argmax_p scores[bh][p][q] (first max, like jnp.argmax)
// Consecutive threads = consecutive q -> coalesced 128B rows per p step.
// ---------------------------------------------------------------------------
__global__ void argmax_kernel(const float* __restrict__ scores,
                              int* __restrict__ idx) {
  int t = blockIdx.x * blockDim.x + threadIdx.x;
  if (t >= 32 * 900) return;
  int bh = t / 900, q = t - bh * 900;
  const float* sb = scores + (size_t)bh * 810000 + q;
  float best = sb[0];
  int bi = 0;
  for (int p = 1; p < 900; ++p) {
    if (p + 16 < 900) __builtin_prefetch(sb + (size_t)(p + 16) * 900, 0, 0);
    float v = sb[(size_t)p * 900];
    if (v > best) { best = v; bi = p; }
  }
  idx[t] = bi;
}

// ---------------------------------------------------------------------------
// Kernel 4: fused gaussian modulation + scale(1/8) + softmax over q (900).
// One block per (bh, p) row; row kept in LDS between passes.
// gauss = exp(-((xs[x0]-idx%30)^2 + (ys[y0]-idx/30)^2) / (2*sigma^2)), s=5.
// ---------------------------------------------------------------------------
__global__ __launch_bounds__(256) void gsoftmax_kernel(
    float* __restrict__ scores, const int* __restrict__ idx) {
  __shared__ float srow[900];
  __shared__ float red[256];
  int p = blockIdx.x, bh = blockIdx.y, tid = threadIdx.x;
  float* row = scores + (size_t)bh * 810000 + (size_t)p * 900;
  const int* irow = idx + bh * 900;
  float ys = -1.f + (2.f / 29.f) * (float)(p / 30);
  float xs = -1.f + (2.f / 29.f) * (float)(p % 30);
  float lmax = -1e30f;
  for (int q = tid; q < 900; q += 256) {
    int id = irow[q];
    float fy = (float)(id / 30), fx = (float)(id % 30);
    float dx = xs - fx, dy = ys - fy;
    float g = __expf(-(dx * dx + dy * dy) * 0.02f);  // 1/(2*25)
    float s = row[q] * g * 0.125f;                   // 1/sqrt(64)
    srow[q] = s;
    lmax = fmaxf(lmax, s);
  }
  red[tid] = lmax;
  __syncthreads();
  for (int off = 128; off > 0; off >>= 1) {
    if (tid < off) red[tid] = fmaxf(red[tid], red[tid + off]);
    __syncthreads();
  }
  float mx = red[0];
  __syncthreads();
  float lsum = 0.f;
  for (int q = tid; q < 900; q += 256) {
    float e = __expf(srow[q] - mx);
    srow[q] = e;
    lsum += e;
  }
  red[tid] = lsum;
  __syncthreads();
  for (int off = 128; off > 0; off >>= 1) {
    if (tid < off) red[tid] += red[tid + off];
    __syncthreads();
  }
  float inv = 1.f / red[0];
  for (int q = tid; q < 900; q += 256) row[q] = srow[q] * inv;
}

// ---------------------------------------------------------------------------
// Kernel 5: conv1 3x3 SAME, 8->32 ch, ReLU. attn [4,8,900,900] f32 -> mid f16.
// 16x16 pixel tile staged in LDS with halo; weights in LDS.
// (Halo needs real zeros at image border, so no async/clamp trick here.)
// ---------------------------------------------------------------------------
__global__ __launch_bounds__(256) void conv1_kernel(
    const float* __restrict__ attn, const float* __restrict__ w,
    const float* __restrict__ bias, _Float16* __restrict__ mid) {
  __shared__ float wgt[32 * 8 * 9];
  __shared__ float tile[8][18][18];
  int tx = threadIdx.x, ty = threadIdx.y;
  int tid = ty * 16 + tx;
  int bx = blockIdx.x * 16, by = blockIdx.y * 16, b = blockIdx.z;
  for (int i = tid; i < 32 * 8 * 9; i += 256) wgt[i] = w[i];
  for (int i = tid; i < 8 * 18 * 18; i += 256) {
    int ci = i / 324, rem = i - ci * 324;
    int r = rem / 18, c = rem - r * 18;
    int iy = by + r - 1, ix = bx + c - 1;
    float v = 0.f;
    if (iy >= 0 && iy < 900 && ix >= 0 && ix < 900)
      v = attn[(size_t)(b * 8 + ci) * 810000 + (size_t)iy * 900 + ix];
    tile[ci][r][c] = v;
  }
  __syncthreads();
  int y = by + ty, x = bx + tx;
  if (y >= 900 || x >= 900) return;
  size_t obase = (size_t)b * 32 * 810000 + (size_t)y * 900 + x;
  for (int oc = 0; oc < 32; ++oc) {
    float acc = bias[oc];
#pragma unroll
    for (int ci = 0; ci < 8; ++ci)
#pragma unroll
      for (int ky = 0; ky < 3; ++ky)
#pragma unroll
        for (int kx = 0; kx < 3; ++kx)
          acc += tile[ci][ty + ky][tx + kx] *
                 wgt[((oc * 8 + ci) * 3 + ky) * 3 + kx];
    mid[obase + (size_t)oc * 810000] = (_Float16)fmaxf(acc, 0.f);
  }
}

// ---------------------------------------------------------------------------
// Kernel 6: fused conv2 (32->8, 3x3, ReLU) + sum_q out*value[b,q].
// Accumulates into xacc[b][co][y] via shuffle row-reduce + one atomic per
// (row, co) per tile -> never materializes the second [4,8,900,900] tensor.
// ---------------------------------------------------------------------------
__global__ __launch_bounds__(256) void conv2_reduce_kernel(
    const _Float16* __restrict__ mid, const float* __restrict__ w,
    const float* __restrict__ bias, const float* __restrict__ value,
    float* __restrict__ xacc) {
  __shared__ float wgt[8 * 32 * 9];
  __shared__ _Float16 tile[32][18][18];
  int tx = threadIdx.x, ty = threadIdx.y;
  int tid = ty * 16 + tx;
  int bx = blockIdx.x * 16, by = blockIdx.y * 16, b = blockIdx.z;
  for (int i = tid; i < 8 * 32 * 9; i += 256) wgt[i] = w[i];
  for (int i = tid; i < 32 * 18 * 18; i += 256) {
    int ci = i / 324, rem = i - ci * 324;
    int r = rem / 18, c = rem - r * 18;
    int iy = by + r - 1, ix = bx + c - 1;
    _Float16 v = (_Float16)0.f;
    if (iy >= 0 && iy < 900 && ix >= 0 && ix < 900)
      v = mid[(size_t)(b * 32 + ci) * 810000 + (size_t)iy * 900 + ix];
    tile[ci][r][c] = v;
  }
  __syncthreads();
  int y = by + ty, x = bx + tx;
  bool valid = (y < 900) && (x < 900);
  float vv = valid ? value[b * 900 + x] : 0.f;
  for (int co = 0; co < 8; ++co) {
    float acc = bias[co];
    for (int ci = 0; ci < 32; ++ci)
#pragma unroll
      for (int ky = 0; ky < 3; ++ky)
#pragma unroll
        for (int kx = 0; kx < 3; ++kx)
          acc += (float)tile[ci][ty + ky][tx + kx] *
                 wgt[((co * 32 + ci) * 3 + ky) * 3 + kx];
    float part = valid ? fmaxf(acc, 0.f) * vv : 0.f;
    // reduce across the 16 x-lanes (xor < 16 stays within same ty half-wave)
    part += __shfl_xor(part, 8, 32);
    part += __shfl_xor(part, 4, 32);
    part += __shfl_xor(part, 2, 32);
    part += __shfl_xor(part, 1, 32);
    if (tx == 0 && y < 900)
      atomicAdd(&xacc[(size_t)(b * 8 + co) * 900 + y], part);
  }
}

__global__ void zero_kernel(float* __restrict__ p, int n) {
  int t = blockIdx.x * blockDim.x + threadIdx.x;
  if (t < n) p[t] = 0.f;
}

// out[b][p] = mean_h xacc[b][h][p]
__global__ void finalize_kernel(const float* __restrict__ xacc,
                                float* __restrict__ out) {
  int t = blockIdx.x * blockDim.x + threadIdx.x;
  if (t >= 3600) return;
  int b = t / 900, p = t - b * 900;
  float s = 0.f;
#pragma unroll
  for (int h = 0; h < 8; ++h) s += xacc[(size_t)(b * 8 + h) * 900 + p];
  out[t] = s * 0.125f;
}

// ---------------------------------------------------------------------------
extern "C" void kernel_launch(void* const* d_in, const int* in_sizes, int n_in,
                              void* d_out, int out_size, void* d_ws,
                              size_t ws_size, hipStream_t stream) {
  const float* query = (const float*)d_in[0];
  const float* key_t = (const float*)d_in[1];
  const float* value = (const float*)d_in[2];
  const float* Wq    = (const float*)d_in[3];
  const float* bq    = (const float*)d_in[4];
  const float* Wk    = (const float*)d_in[5];
  const float* bk    = (const float*)d_in[6];
  const float* c1w   = (const float*)d_in[7];
  const float* c1b   = (const float*)d_in[8];
  const float* c2w   = (const float*)d_in[9];
  const float* c2b   = (const float*)d_in[10];
  float* out = (float*)d_out;

  // Workspace layout (bytes):
  //   qh f16  [4,8,900,64]      @ 0          (3,686,400)
  //   kh f16  [4,8,900,64]      @ 3,686,400  (3,686,400)
  //   scores f32 [32,900,900]   @ 7,372,800  (103,680,000)  (reused as attn)
  //   mid f16 [4,32,900,900]    @ 111,052,800 (207,360,000)
  //   idx i32 [32,900]          @ 318,412,800 (115,200)
  //   xacc f32 [4,8,900]        @ 318,528,000 (115,200)
  char* ws = (char*)d_ws;
  _Float16* qh     = (_Float16*)(ws);
  _Float16* kh     = (_Float16*)(ws + 3686400);
  float*    scores = (float*)   (ws + 7372800);
  _Float16* mid    = (_Float16*)(ws + 111052800);
  int*      idx    = (int*)     (ws + 318412800);
  float*    xacc   = (float*)   (ws + 318528000);

  proj_kernel<<<dim3(225, 4), 256, 0, stream>>>(query, Wq, bq, qh);
  proj_kernel<<<dim3(225, 4), 256, 0, stream>>>(key_t, Wk, bk, kh);
  scores_kernel<<<dim3(29, 15, 32), 256, 0, stream>>>(qh, kh, scores);
  argmax_kernel<<<dim3(113), 256, 0, stream>>>(scores, idx);
  gsoftmax_kernel<<<dim3(900, 32), 256, 0, stream>>>(scores, idx);
  conv1_kernel<<<dim3(57, 57, 4), dim3(16, 16), 0, stream>>>(scores, c1w, c1b,
                                                             mid);
  zero_kernel<<<dim3(113), 256, 0, stream>>>(xacc, 28800);
  conv2_reduce_kernel<<<dim3(57, 57, 4), dim3(16, 16), 0, stream>>>(
      mid, c2w, c2b, value, xacc);
  finalize_kernel<<<dim3(15), 256, 0, stream>>>(xacc, out);
}
